// RevViT_GLOW_79517024518921
// MI455X (gfx1250) — matrix operand order repaired and checked by reading, not compiled, not run on hardware
//
#include <hip/hip_runtime.h>
#include <hip/hip_bf16.h>
#include <math.h>

// ---------------------------------------------------------------------------
// Types for CDNA5 WMMA (wave32): A/B 16x32 bf16 fragments, C/D 16x16 f32.
// ---------------------------------------------------------------------------
typedef __attribute__((ext_vector_type(16))) __bf16 v16bf;
typedef __attribute__((ext_vector_type(8)))  float  v8f;
typedef float f4 __attribute__((ext_vector_type(4), aligned(4)));

#define HALF_LOG2PI 0.918938533204672741f

// ---------------------------------------------------------------------------
// Generic batched WMMA GEMM:  C = alpha * A @ B (+bias) (+relu)
//   A: fp32, row-major, row stride lda, batch stride sA           (M x K)
//   B: fp32, btrans=0: (K x N) row stride ldb; btrans=1: (N x K) row stride ldb
//   C: fp32, row stride ldc, batch stride sC                      (M x N)
// One wave computes a 16x32 C strip (two 16x16 tiles sharing one A fragment).
// K is split into a branch-free software-pipelined fast loop (b128 loads) for
// interior tiles plus a single guarded tail; edge tiles use address-clamped
// unpredicated loads. EXEC is all-ones at every WMMA (wave-uniform early out).
// ---------------------------------------------------------------------------
__global__ __launch_bounds__(128) void k_wmma_gemm(
    const float* __restrict__ A, long lda, long sA,
    const float* __restrict__ B, long ldb, long sB, int btrans,
    float* __restrict__ C, long ldc, long sC,
    int M, int N, int K, float alpha,
    const float* __restrict__ bias, int relu, int nbatch)
{
  long wave = (long)blockIdx.x * 4 + (threadIdx.x >> 5);
  int  lane = threadIdx.x & 31;
  int  tiles_m  = (M + 15) >> 4;
  int  tiles_n  = (N + 15) >> 4;
  int  tiles_np = (tiles_n + 1) >> 1;          // N-tile pairs (32-wide strips)
  long tiles    = (long)tiles_m * tiles_np;
  if (wave >= tiles * nbatch) return;          // uniform per wave
  int  batch = (int)(wave / tiles);
  long t     = wave - (long)batch * tiles;
  int  tm    = (int)(t / tiles_np) << 4;
  int  tn    = (int)(t % tiles_np) << 5;

  const float* Ab = A + (long)batch * sA;
  const float* Bb = B + (long)batch * sB;
  float*       Cb = C + (long)batch * sC;

  int hl = lane & 15;           // row within A-tile / col within B-tile
  int g  = lane >> 4;           // lane group (K half: K = g*16 + j)
  int row  = tm + hl;
  int col0 = tn + hl;
  int col1 = tn + 16 + hl;
  bool aFull  = (tm + 16 <= M);
  bool bFull  = (tn + 32 <= N);
  int rc  = (row  < M) ? row  : (M - 1);       // clamped (always-legal) indices
  int cc0 = (col0 < N) ? col0 : (N - 1);
  int cc1 = (col1 < N) ? col1 : (N - 1);

  // ---- fragment builders -------------------------------------------------
  // A 16x32: K = g*16+j
  auto loadA_fast = [&](int k0) -> v16bf {
    v16bf af;
    const float* ap = Ab + (long)row * lda + k0 + g * 16;
    f4 x0 = *(const f4*)(ap);
    f4 x1 = *(const f4*)(ap + 4);
    f4 x2 = *(const f4*)(ap + 8);
    f4 x3 = *(const f4*)(ap + 12);
#pragma unroll
    for (int j = 0; j < 4; ++j) {
      af[j]      = (__bf16)x0[j];
      af[4 + j]  = (__bf16)x1[j];
      af[8 + j]  = (__bf16)x2[j];
      af[12 + j] = (__bf16)x3[j];
    }
    return af;
  };
  auto loadA_guard = [&](int k0) -> v16bf {
    v16bf af;
#pragma unroll
    for (int j = 0; j < 16; ++j) {
      int kk = k0 + g * 16 + j;
      int kc = (kk < K) ? kk : (K - 1);
      float av = Ab[(long)rc * lda + kc];
      af[j] = (row < M && kk < K) ? (__bf16)av : (__bf16)0.f;
    }
    return af;
  };
  // B 32x16: lane holds column, K = k0 + g*16 + j
  auto loadB_fast = [&](int col, int k0) -> v16bf {
    v16bf bf;
    if (btrans) {
      const float* bp = Bb + (long)col * ldb + k0 + g * 16;
      f4 x0 = *(const f4*)(bp);
      f4 x1 = *(const f4*)(bp + 4);
      f4 x2 = *(const f4*)(bp + 8);
      f4 x3 = *(const f4*)(bp + 12);
#pragma unroll
      for (int j = 0; j < 4; ++j) {
        bf[j]      = (__bf16)x0[j];
        bf[4 + j]  = (__bf16)x1[j];
        bf[8 + j]  = (__bf16)x2[j];
        bf[12 + j] = (__bf16)x3[j];
      }
    } else {
      const float* bp = Bb + (long)(k0 + g * 16) * ldb + col;
#pragma unroll
      for (int j = 0; j < 16; ++j)
        bf[j] = (__bf16)bp[(long)j * ldb];
    }
    return bf;
  };
  auto loadB_guard = [&](int col, int cc, int k0) -> v16bf {
    v16bf bf;
#pragma unroll
    for (int j = 0; j < 16; ++j) {
      int kk = k0 + g * 16 + j;
      int kc = (kk < K) ? kk : (K - 1);
      float bv = btrans ? Bb[(long)cc * ldb + kc] : Bb[(long)kc * ldb + cc];
      bf[j] = (col < N && kk < K) ? (__bf16)bv : (__bf16)0.f;
    }
    return bf;
  };

  v8f acc0 = {0.f, 0.f, 0.f, 0.f, 0.f, 0.f, 0.f, 0.f};
  v8f acc1 = {0.f, 0.f, 0.f, 0.f, 0.f, 0.f, 0.f, 0.f};
  const int kfull = K & ~31;

  if (aFull && bFull) {
    // ---- branch-free fast loop, 2-stage software pipeline ----------------
    if (kfull > 0) {
      v16bf af = loadA_fast(0);
      v16bf b0 = loadB_fast(col0, 0);
      v16bf b1 = loadB_fast(col1, 0);
      for (int k0 = 32; k0 < kfull; k0 += 32) {
        v16bf afn = loadA_fast(k0);
        v16bf b0n = loadB_fast(col0, k0);
        v16bf b1n = loadB_fast(col1, k0);
        acc0 = __builtin_amdgcn_wmma_f32_16x16x32_bf16(false, af, false, b0,
                                                       (short)0, acc0, false, false);
        acc1 = __builtin_amdgcn_wmma_f32_16x16x32_bf16(false, af, false, b1,
                                                       (short)0, acc1, false, false);
        af = afn; b0 = b0n; b1 = b1n;
      }
      acc0 = __builtin_amdgcn_wmma_f32_16x16x32_bf16(false, af, false, b0,
                                                     (short)0, acc0, false, false);
      acc1 = __builtin_amdgcn_wmma_f32_16x16x32_bf16(false, af, false, b1,
                                                     (short)0, acc1, false, false);
    }
  } else {
    // ---- edge tiles: address-clamped, unpredicated loads -----------------
    for (int k0 = 0; k0 < kfull; k0 += 32) {
      v16bf af = loadA_guard(k0);
      v16bf b0 = loadB_guard(col0, cc0, k0);
      v16bf b1 = loadB_guard(col1, cc1, k0);
      acc0 = __builtin_amdgcn_wmma_f32_16x16x32_bf16(false, af, false, b0,
                                                     (short)0, acc0, false, false);
      acc1 = __builtin_amdgcn_wmma_f32_16x16x32_bf16(false, af, false, b1,
                                                     (short)0, acc1, false, false);
    }
  }
  if (kfull < K) {
    // ---- single guarded K-tail -------------------------------------------
    v16bf af = loadA_guard(kfull);
    v16bf b0 = loadB_guard(col0, cc0, kfull);
    v16bf b1 = loadB_guard(col1, cc1, kfull);
    acc0 = __builtin_amdgcn_wmma_f32_16x16x32_bf16(false, af, false, b0,
                                                   (short)0, acc0, false, false);
    acc1 = __builtin_amdgcn_wmma_f32_16x16x32_bf16(false, af, false, b1,
                                                   (short)0, acc1, false, false);
  }

  // C layout: VGPR i -> M = tm + i + 8*g, N = col
  if (col0 < N) {
    float bv = bias ? bias[col0] : 0.f;
#pragma unroll
    for (int i = 0; i < 8; ++i) {
      int r = tm + i + 8 * g;
      if (r < M) {
        float v = acc0[i] * alpha + bv;
        if (relu) v = fmaxf(v, 0.f);
        Cb[(long)r * ldc + col0] = v;
      }
    }
  }
  if (col1 < N) {
    float bv = bias ? bias[col1] : 0.f;
#pragma unroll
    for (int i = 0; i < 8; ++i) {
      int r = tm + i + 8 * g;
      if (r < M) {
        float v = acc1[i] * alpha + bv;
        if (relu) v = fmaxf(v, 0.f);
        Cb[(long)r * ldc + col1] = v;
      }
    }
  }
}

// ---------------------------------------------------------------------------
// LayerNorm: one wave32 per row, shuffle reductions.
// Row r -> in  + (r/rpb)*inBS  + (r%rpb)*width
//          out + (r/rpb)*outBS + (r%rpb)*width
// ---------------------------------------------------------------------------
__global__ __launch_bounds__(128) void k_layernorm(
    const float* __restrict__ in, long inBS,
    float* __restrict__ out, long outBS,
    int rpb, int width,
    const float* __restrict__ w, const float* __restrict__ b, int rows)
{
  int wid  = blockIdx.x * 4 + (threadIdx.x >> 5);
  int lane = threadIdx.x & 31;
  if (wid >= rows) return;
  const float* x = in  + (long)(wid / rpb) * inBS  + (long)(wid % rpb) * width;
  float*       y = out + (long)(wid / rpb) * outBS + (long)(wid % rpb) * width;

  float s = 0.f;
  for (int c = lane; c < width; c += 32) s += x[c];
  for (int o = 16; o > 0; o >>= 1) s += __shfl_xor(s, o, 32);
  float mean = s / (float)width;

  float v = 0.f;
  for (int c = lane; c < width; c += 32) { float d = x[c] - mean; v += d * d; }
  for (int o = 16; o > 0; o >>= 1) v += __shfl_xor(v, o, 32);
  float rstd = rsqrtf(v / (float)width + 1e-6f);

  for (int c = lane; c < width; c += 32)
    y[c] = (x[c] - mean) * rstd * w[c] + b[c];
}

// ---------------------------------------------------------------------------
// Row softmax (in place), one wave per row.
// ---------------------------------------------------------------------------
__global__ __launch_bounds__(128) void k_softmax(float* __restrict__ buf,
                                                 int width, int rows)
{
  int wid  = blockIdx.x * 4 + (threadIdx.x >> 5);
  int lane = threadIdx.x & 31;
  if (wid >= rows) return;
  float* x = buf + (long)wid * width;

  float m = -3.4e38f;
  for (int c = lane; c < width; c += 32) m = fmaxf(m, x[c]);
  for (int o = 16; o > 0; o >>= 1) m = fmaxf(m, __shfl_xor(m, o, 32));

  float s = 0.f;
  for (int c = lane; c < width; c += 32) { float e = __expf(x[c] - m); x[c] = e; s += e; }
  for (int o = 16; o > 0; o >>= 1) s += __shfl_xor(s, o, 32);
  float inv = 1.f / s;
  for (int c = lane; c < width; c += 32) x[c] *= inv;
}

// ---------------------------------------------------------------------------
// Elementwise helpers
// ---------------------------------------------------------------------------
__global__ void k_zero(float* __restrict__ p, int n)
{
  int i = blockIdx.x * blockDim.x + threadIdx.x;
  if (i < n) p[i] = 0.f;
}

// patches[m,k]: m=b*576+ph*24+pw, k=c*256+py*16+px  (IMG=384, PATCH=16)
__global__ void k_patchify(const float* __restrict__ img, float* __restrict__ outp,
                           long total)
{
  long idx = (long)blockIdx.x * blockDim.x + threadIdx.x;
  if (idx >= total) return;
  int  k = (int)(idx % 768);
  long m = idx / 768;
  int  p = (int)(m % 576);
  int  b = (int)(m / 576);
  int ph = p / 24, pw = p % 24;
  int c = k >> 8, r = k & 255, py = r >> 4, px = r & 15;
  outp[idx] = img[(((long)b * 3 + c) * 384 + ph * 16 + py) * 384 + pw * 16 + px];
}

// x[b,0,:]=cls+pos[0]; x[b,t,:]=X0[b,t-1,:]+pos[t]
__global__ void k_tokens(const float* __restrict__ x0, const float* __restrict__ cls,
                         const float* __restrict__ pos, float* __restrict__ xo,
                         long total)
{
  long idx = (long)blockIdx.x * blockDim.x + threadIdx.x;
  if (idx >= total) return;
  int  c = (int)(idx % 768);
  long r = idx / 768;
  int  t = (int)(r % 577);
  long b = r / 577;
  float v = (t == 0) ? cls[c] : x0[((long)b * 576 + (t - 1)) * 768 + c];
  xo[idx] = v + pos[(long)t * 768 + c];
}

// sq[b, 2p+s, c] = x[b, p, 2c+s]  (channel de-interleave squeeze)
__global__ void k_squeeze(const float* __restrict__ x, long xBS,
                          float* __restrict__ sq, int P, int d, long total)
{
  long idx = (long)blockIdx.x * blockDim.x + threadIdx.x;
  if (idx >= total) return;
  int  c = (int)(idx % d);
  long jj = idx / d;
  int  j = (int)(jj % (2 * P));
  long b = jj / (2 * P);
  sq[idx] = x[b * xBS + (long)(j >> 1) * (2 * d) + 2 * c + (j & 1)];
}

// o = a + b with independent batch strides (width-contiguous rows)
__global__ void k_add_rows(const float* __restrict__ a, long aBS,
                           const float* __restrict__ bsrc, long bBS,
                           float* __restrict__ o, long oBS,
                           int rpb, int width, long total)
{
  long idx = (long)blockIdx.x * blockDim.x + threadIdx.x;
  if (idx >= total) return;
  int  c = (int)(idx % width);
  long r = idx / width;
  int  t = (int)(r % rpb);
  long n = r / rpb;
  o[n * oBS + (long)t * width + c] =
      a[n * aBS + (long)t * width + c] + bsrc[n * bBS + (long)t * width + c];
}

// Gaussian log-density sum vs per-element prior rows [mean(0..d) | log_sd(d..2d)]
__global__ __launch_bounds__(128) void k_logp(
    const float* __restrict__ z, long zBS, const float* __restrict__ prior,
    int rpb, int d, float* __restrict__ out, int rows)
{
  int r = blockIdx.x;
  if (r >= rows) return;
  int n = r / rpb, t = r % rpb;
  const float* zr = z + (long)n * zBS + (long)t * d;
  const float* pr = prior + (long)r * 2 * d;
  float s = 0.f;
  for (int c = threadIdx.x; c < d; c += blockDim.x) {
    float mean = pr[c], lsd = pr[d + c];
    float df = zr[c] - mean;
    s += -HALF_LOG2PI - lsd - 0.5f * df * df * __expf(-2.f * lsd);
  }
  __shared__ float red[128];
  red[threadIdx.x] = s;
  __syncthreads();
  for (int o = 64; o > 0; o >>= 1) {
    if (threadIdx.x < o) red[threadIdx.x] += red[threadIdx.x + o];
    __syncthreads();
  }
  if (threadIdx.x == 0) atomicAdd(&out[n], red[0]);
}

// Same but with a constant mean/log_sd vector (final no-split block)
__global__ __launch_bounds__(128) void k_logp_const(
    const float* __restrict__ y, const float* __restrict__ ml, int d,
    float* __restrict__ out, int rpb, int rows)
{
  int r = blockIdx.x;
  if (r >= rows) return;
  int n = r / rpb;
  const float* yr = y + (long)r * d;
  float s = 0.f;
  for (int c = threadIdx.x; c < d; c += blockDim.x) {
    float mean = ml[c], lsd = ml[d + c];
    float df = yr[c] - mean;
    s += -HALF_LOG2PI - lsd - 0.5f * df * df * __expf(-2.f * lsd);
  }
  __shared__ float red[128];
  red[threadIdx.x] = s;
  __syncthreads();
  for (int o = 64; o > 0; o >>= 1) {
    if (threadIdx.x < o) red[threadIdx.x] += red[threadIdx.x + o];
    __syncthreads();
  }
  if (threadIdx.x == 0) atomicAdd(&out[n], red[0]);
}

// LN(zeros) == norm_b, so prior on zeros is a constant vector:
// pr[j] = sum_c norm_b[c]*prior_w[c,j] + prior_b[j]
__global__ void k_prior_const(const float* __restrict__ nb,
                              const float* __restrict__ pw,
                              const float* __restrict__ pb,
                              float* __restrict__ outp, int d)
{
  int j = blockIdx.x * blockDim.x + threadIdx.x;
  if (j >= 2 * d) return;
  float s = pb[j];
  for (int c = 0; c < d; ++c) s += nb[c] * pw[(long)c * 2 * d + j];
  outp[j] = s;
}

// ---------------------------------------------------------------------------
// Host orchestration
// ---------------------------------------------------------------------------
static void launch_gemm(hipStream_t s, const float* A, long lda, long sA,
                        const float* B, long ldb, long sB, int bt,
                        float* C, long ldc, long sC,
                        int M, int N, int K, float alpha,
                        const float* bias, int relu, int nb)
{
  long tiles_m  = (M + 15) / 16;
  long tiles_np = ((N + 15) / 16 + 1) / 2;
  long waves = tiles_m * tiles_np * nb;
  int blocks = (int)((waves + 3) / 4);
  k_wmma_gemm<<<blocks, 128, 0, s>>>(A, lda, sA, B, ldb, sB, bt,
                                     C, ldc, sC, M, N, K, alpha, bias, relu, nb);
}

static void launch_ln(hipStream_t s, const float* in, long inBS,
                      float* out, long outBS, int rpb, int width,
                      const float* w, const float* b, int rows)
{
  k_layernorm<<<(rows + 3) / 4, 128, 0, s>>>(in, inBS, out, outBS, rpb, width, w, b, rows);
}

extern "C" void kernel_launch(void* const* d_in, const int* in_sizes, int n_in,
                              void* d_out, int out_size, void* d_ws, size_t ws_size,
                              hipStream_t stream)
{
  (void)in_sizes; (void)n_in; (void)out_size; (void)ws_size;

  const float* img     = (const float*)d_in[0];
  const float* patch_w = (const float*)d_in[1];
  const float* patch_b = (const float*)d_in[2];
  const float* cls     = (const float*)d_in[3];
  const float* pos     = (const float*)d_in[4];
  // per-block weights: 20 tensors per block, dict order
  #define BW(j) ((const float*)d_in[5 + i * 20 + (j)])
  enum { W_N1W = 0, W_N1B, W_QKVW, W_QKVB, W_PRJW, W_PRJB, W_N2W, W_N2B,
         W_F1W, W_F1B, W_F2W, W_F2B, W_NW, W_NB, W_PRW, W_PRB,
         W_HNW, W_HNB, W_HDW, W_HDB };

  float* ws  = (float*)d_ws;
  float* out = (float*)d_out;

  const int  Bn = 32, P = 577, H = 12;
  const long NE = 14174208L;                 // 18456*768 elements

  float* bx    = ws;                          // tokens / block-0 x       (NE)
  float* bsq   = bx    + NE;                  // squeeze / patch staging  (NE)
  float* bbig  = bsq   + NE;                  // qkv / fc1 / prior        (2*NE)
  float* btmp  = bbig  + 2 * NE;              // ln1 / attn out / ln(y1)  (NE/2)
  float* btmp2 = btmp  + NE / 2;              // proj / fc2 out           (NE/2)
  float* by    = btmp2 + NE / 2;              // y = [y1 | y2]            (NE)
  float* bsc   = by    + NE;                  // per-image scores (12*577*577)
  float* bcst  = bsc   + 12L * 577 * 577;     // const prior (96) ...
  float* byn   = bcst  + 256;                 // ... + head LN rows (32*48)

  // 0) zero log_p accumulators (d_out[0..32) — accumulated via atomics)
  k_zero<<<1, 64, 0, stream>>>(out, 32);

  // 1) patchify + patch embed GEMM: X0 = patches @ patch_w^T + patch_b
  {
    long tot = (long)Bn * 576 * 768;
    k_patchify<<<(int)((tot + 255) / 256), 256, 0, stream>>>(img, bsq, tot);
    launch_gemm(stream, bsq, 768, 0, patch_w, 768, 0, /*bt=*/1,
                bbig, 768, 0, Bn * 576, 768, 768, 1.f, patch_b, 0, 1);
    long tot2 = (long)Bn * P * 768;
    k_tokens<<<(int)((tot2 + 255) / 256), 256, 0, stream>>>(bbig, cls, pos, bx, tot2);
  }

  const float* xsrc = bx;
  long xBS = (long)P * 768;
  int  E   = 768;

  for (int i = 0; i < 4; ++i) {
    const int d  = E >> 1;
    const int hd = d / H;
    const int dh = 4 * d;
    const long M    = (long)Bn * P;       // rows of flattened activations
    const long sqT  = (long)2 * P * d;    // batch stride of (n,2P,d) buffers
    const long rowT = (long)P * d;        // batch stride of (n,P,d) buffers
    const long tot  = M * d;

    // squeeze -> bsq (n, 2P, d); x2 = tokens [0,P), x1 = tokens [P,2P)
    k_squeeze<<<(int)((2 * tot + 255) / 256), 256, 0, stream>>>(
        xsrc, xBS, bsq, P, d, 2 * tot);

    // ln1 = LN(x1) -> btmp (n,P,d)
    launch_ln(stream, bsq + (long)P * d, sqT, btmp, rowT, P, d,
              BW(W_N1W), BW(W_N1B), (int)M);

    // qkv = ln1 @ qkv_w + qkv_b -> bbig (n,P,3d)
    launch_gemm(stream, btmp, d, 0, BW(W_QKVW), 3 * d, 0, 0,
                bbig, 3 * d, 0, (int)M, 3 * d, d, 1.f, BW(W_QKVB), 0, 1);

    // attention per image, 12-head batched GEMMs
    const float scale = 1.f / sqrtf((float)hd);
    for (int n = 0; n < Bn; ++n) {
      const float* qb = bbig + (long)n * P * 3 * d;
      // scores[h] = q @ k^T * scale
      launch_gemm(stream, qb, 3 * d, hd, qb + d, 3 * d, hd, /*bt=*/1,
                  bsc, P, (long)P * P, P, P, hd, scale, nullptr, 0, H);
      k_softmax<<<(H * P + 3) / 4, 128, 0, stream>>>(bsc, P, H * P);
      // out[h] = probs @ v, stored as (t, h*hd+e) into btmp (n,P,d)
      launch_gemm(stream, bsc, P, (long)P * P, qb + 2 * d, 3 * d, hd, 0,
                  btmp + (long)n * P * d, d, hd, P, hd, P, 1.f, nullptr, 0, H);
    }

    // proj -> btmp2 (n,P,d)
    launch_gemm(stream, btmp, d, 0, BW(W_PRJW), d, 0, 0,
                btmp2, d, 0, (int)M, d, d, 1.f, BW(W_PRJB), 0, 1);

    // y2 = x2 + attn -> by tokens [P,2P)
    k_add_rows<<<(int)((tot + 255) / 256), 256, 0, stream>>>(
        bsq, sqT, btmp2, rowT, by + (long)P * d, sqT, P, d, tot);

    // ln2 = LN(y2) -> btmp
    launch_ln(stream, by + (long)P * d, sqT, btmp, rowT, P, d,
              BW(W_N2W), BW(W_N2B), (int)M);

    // mlp: fc1 (+relu) -> bbig ; fc2 -> btmp2
    launch_gemm(stream, btmp, d, 0, BW(W_F1W), dh, 0, 0,
                bbig, dh, 0, (int)M, dh, d, 1.f, BW(W_F1B), 1, 1);
    launch_gemm(stream, bbig, dh, 0, BW(W_F2W), d, 0, 0,
                btmp2, d, 0, (int)M, d, dh, 1.f, BW(W_F2B), 0, 1);

    // y1 = x1 + mlp -> by tokens [0,P)
    k_add_rows<<<(int)((tot + 255) / 256), 256, 0, stream>>>(
        bsq + (long)P * d, sqT, btmp2, rowT, by, sqT, P, d, tot);

    if (i < 3) {
      // split: x_next = y1 ; z = y2 ; prior from LN(y1)
      launch_ln(stream, by, sqT, btmp, rowT, P, d, BW(W_NW), BW(W_NB), (int)M);
      launch_gemm(stream, btmp, d, 0, BW(W_PRW), 2 * d, 0, 0,
                  bbig, 2 * d, 0, (int)M, 2 * d, d, 1.f, BW(W_PRB), 0, 1);
      k_logp<<<(int)M, 128, 0, stream>>>(by + (long)P * d, sqT, bbig, P, d, out, (int)M);
      xsrc = by;
      xBS  = sqT;     // y1 = first P tokens, row stride d (=next E)
      E    = d;
    } else {
      // final: constant prior over whole y, head on LN(y[:,0])
      k_prior_const<<<1, 128, 0, stream>>>(BW(W_NB), BW(W_PRW), BW(W_PRB), bcst, d);
      k_logp_const<<<(int)(2 * M), 128, 0, stream>>>(by, bcst, d, out, 2 * P,
                                                     (int)(2 * M));
      launch_ln(stream, by, sqT, byn, d, 1, d, BW(W_HNW), BW(W_HNB), Bn);
      launch_gemm(stream, byn, d, 0, BW(W_HDW), d, 0, 0,
                  out + 32, d, 0, Bn, d, d, 1.f, BW(W_HDB), 0, 1);
    }
  }
  #undef BW
}
